// LinearCrossAttention_42039139893668
// MI455X (gfx1250) — matrix operand order repaired and checked by reading, not compiled, not run on hardware
//
#include <hip/hip_runtime.h>
#include <hip/hip_bf16.h>

// ---------------------------------------------------------------------------
// LinearCrossAttention on MI455X (gfx1250, wave32, WMMA 16x16x32 f16->f32)
//
// y = Wout * blockdiag_h(ctx_h^T / rowsum) * (SCALE*Wq) * x + bout
// where ctx_h = exp(K_h - rowmax) * V_h^T,  K = Wk*context, V = Wv*context.
// Softmax stats (rowmax/rowsum) computed over n = 32768 tokens per row.
//
// Workspace layout (bytes, needs ~64.3 MiB):
//   [0        ) Wk16      32768
//   [32768    ) Wv16      32768
//   [65536    ) Weff16   131072   (4 batches x 128x128 half)
//   [196608   ) rowmax     2048   (4x128 orderable-uint)
//   [198656   ) rowsum     2048   (4x128 f32)
//   [200704   ) ctxacc    65536   (4x4x32x32 f32)
//   [266240   ) xh     33554432   (f16 copy of x)
//   [33820672 ) ch     33554432   (f16 copy of context)
// ---------------------------------------------------------------------------

typedef __attribute__((ext_vector_type(16))) _Float16 v16h;
typedef __attribute__((ext_vector_type(8)))  _Float16 v8h;
typedef __attribute__((ext_vector_type(8)))  float    v8f;
typedef __attribute__((ext_vector_type(4)))  float    f4;
typedef __attribute__((ext_vector_type(4)))  _Float16 h4;

#define HEADS 4
#define DIMH  32
#define CCH   128            // channels (== hidden)
#define NTOK  32768          // 32*32*32 tokens
#define BATCH 4
#define SCALEQ 0.17677669529663689f  // 32^-0.5

#define TN   64              // token tile for K/V passes
#define TN2  128             // token tile for final y GEMM
#define LDC  136             // [token][channel] LDS pitch in halves (pad vs bank conflicts)
#define LDK  72              // [row][token] LDS pitch in halves for eK/V tiles

// ---------------- WMMA fragment helpers (assumed CDNA5 wave32 layouts) ------
// A (16x32 f16): lane<16 -> M=lane, halves 0..7 = K[k0..k0+7], 8..15 = K[k0+16..k0+23]
//                lane>=16 -> M=lane-16, K base shifted by +8 (per ISA A-matrix table)
__device__ __forceinline__ v16h load_a16(const _Float16* W, int ld, int m0, int k0, int lane) {
  const int row = m0 + (lane & 15);
  const int kb  = k0 + ((lane & 16) ? 8 : 0);
  const _Float16* p = W + row * ld + kb;
  v8h lo = *reinterpret_cast<const v8h*>(p);
  v8h hi = *reinterpret_cast<const v8h*>(p + 16);
  return __builtin_shufflevector(lo, hi, 0,1,2,3,4,5,6,7,8,9,10,11,12,13,14,15);
}
// B (32x16 f16): lane<16 -> N=lane, halves 0..7 = K[0..7], 8..15 = K[16..23]; lane>=16 -> K base +8.
// Source array is [N-row][K-col] row-major (i.e. B = array^T) -> two 16B runs per lane.
__device__ __forceinline__ v16h load_b16(const _Float16* X, int ld, int n0, int k0, int lane) {
  const _Float16* p = X + (n0 + (lane & 15)) * ld + k0 + ((lane & 16) ? 8 : 0);
  v8h lo = *reinterpret_cast<const v8h*>(p);
  v8h hi = *reinterpret_cast<const v8h*>(p + 16);
  return __builtin_shufflevector(lo, hi, 0,1,2,3,4,5,6,7,8,9,10,11,12,13,14,15);
}
__device__ __forceinline__ v8f wmma_f16(v16h a, v16h b, v8f c) {
  return __builtin_amdgcn_wmma_f32_16x16x32_f16(false, a, false, b, (short)0, c, false, false);
}

// ---------------- misc helpers ---------------------------------------------
__device__ __forceinline__ unsigned fenc(float f) {       // order-preserving f32->u32
  unsigned u = __float_as_uint(f);
  return (u & 0x80000000u) ? ~u : (u | 0x80000000u);
}
__device__ __forceinline__ float fdec(unsigned u) {
  return __uint_as_float((u & 0x80000000u) ? (u ^ 0x80000000u) : ~u);
}
__device__ __forceinline__ float red16_max(float v) {
  #pragma unroll
  for (int m = 8; m; m >>= 1) v = fmaxf(v, __shfl_xor(v, m, 32));
  return v;  // lanes 0-15 and 16-31 reduced independently
}
__device__ __forceinline__ float red16_sum(float v) {
  #pragma unroll
  for (int m = 8; m; m >>= 1) v += __shfl_xor(v, m, 32);
  return v;
}

// stage [128ch][T tokens] f16 global tile -> LDS [token][channel] (pitch LDC)
template<int T>
__device__ __forceinline__ void stage_tile(const _Float16* src, long tile0,
                                           _Float16* lds, int tid) {
  const int c  = tid >> 1;
  const int tq = (tid & 1) * (T / 2);
  const _Float16* gp = src + (long)c * NTOK + tile0 + tq;
  #pragma unroll
  for (int j = 0; j < T / 16; ++j) {
    v8h v = *reinterpret_cast<const v8h*>(gp + j * 8);
    #pragma unroll
    for (int k2 = 0; k2 < 8; ++k2) lds[(tq + j * 8 + k2) * LDC + c] = v[k2];
  }
}

// ---------------- kernel 0a: f32 -> f16 conversion of x and context ---------
__global__ void __launch_bounds__(256)
convert_inputs(const float* __restrict__ x, const float* __restrict__ c,
               _Float16* __restrict__ xh, _Float16* __restrict__ ch) {
  const int total4 = BATCH * CCH * NTOK / 4;   // 4,194,304 float4 groups
  int i = blockIdx.x * blockDim.x + threadIdx.x;
  const int stride = gridDim.x * blockDim.x;
  const f4* xv = reinterpret_cast<const f4*>(x);
  const f4* cv = reinterpret_cast<const f4*>(c);
  h4* xo = reinterpret_cast<h4*>(xh);
  h4* co = reinterpret_cast<h4*>(ch);
  for (; i < total4; i += stride) {
    xo[i] = __builtin_convertvector(xv[i], h4);
    co[i] = __builtin_convertvector(cv[i], h4);
  }
}

// ---------------- kernel 0b: weights -> f16, zero accumulators --------------
__global__ void __launch_bounds__(256)
prep_small(const float* __restrict__ Wk, const float* __restrict__ Wv,
           _Float16* __restrict__ Wk16, _Float16* __restrict__ Wv16,
           unsigned* __restrict__ rmax, float* __restrict__ rsum,
           float* __restrict__ ctxa) {
  const int i = blockIdx.x * 256 + threadIdx.x;   // 64 blocks -> 16384 threads
  Wk16[i] = (_Float16)Wk[i];
  Wv16[i] = (_Float16)Wv[i];
  ctxa[i] = 0.0f;
  if (i < BATCH * CCH) { rmax[i] = 0u; rsum[i] = 0.0f; }
}

// ---------------- kernel 1: K = Wk*ctx tiles, global row-max ----------------
__global__ void __launch_bounds__(256)
kv_rowmax(const _Float16* __restrict__ chg, const _Float16* __restrict__ Wk16,
          unsigned* __restrict__ rmaxg) {
  __shared__ _Float16 ctile[TN * LDC];
  const int b = blockIdx.y;
  const long tile0 = (long)blockIdx.x * TN;
  const int tid = threadIdx.x, lane = tid & 31, wave = tid >> 5;
  stage_tile<TN>(chg + (long)b * CCH * NTOK, tile0, ctile, tid);
  __syncthreads();

  const int m0 = wave * 16;
  const int rofs = (lane & 16) ? 8 : 0;
  v16h afr[4];
  #pragma unroll
  for (int ks = 0; ks < 4; ++ks) afr[ks] = load_a16(Wk16, CCH, m0, ks * 32, lane);

  float rm[8];
  #pragma unroll
  for (int r = 0; r < 8; ++r) rm[r] = -3.0e38f;
  for (int nt = 0; nt < TN / 16; ++nt) {
    v8f acc = {};
    #pragma unroll
    for (int ks = 0; ks < 4; ++ks)
      acc = wmma_f16(afr[ks], load_b16(ctile, LDC, nt * 16, ks * 32, lane), acc);
    #pragma unroll
    for (int r = 0; r < 8; ++r) rm[r] = fmaxf(rm[r], acc[r]);
  }
  #pragma unroll
  for (int r = 0; r < 8; ++r) rm[r] = red16_max(rm[r]);
  if ((lane & 15) == 0) {
    #pragma unroll
    for (int r = 0; r < 8; ++r)
      atomicMax(&rmaxg[b * CCH + m0 + rofs + r], fenc(rm[r]));
  }
}

// ------ kernel 2: recompute K,V; e=exp(k-max); rowsum; ctx += eK * V^T ------
__global__ void __launch_bounds__(256)
kv_softmax_ctx(const _Float16* __restrict__ chg, const _Float16* __restrict__ Wk16,
               const _Float16* __restrict__ Wv16, const unsigned* __restrict__ rmaxg,
               float* __restrict__ rsumg, float* __restrict__ ctxa) {
  __shared__ _Float16 ctile[TN * LDC];
  __shared__ _Float16 kl[CCH * LDK];   // eK  [row 0..127][token 0..TN)
  __shared__ _Float16 vl[CCH * LDK];   // V   [row 0..127][token 0..TN)
  const int b = blockIdx.y;
  const long tile0 = (long)blockIdx.x * TN;
  const int tid = threadIdx.x, lane = tid & 31, wave = tid >> 5;
  stage_tile<TN>(chg + (long)b * CCH * NTOK, tile0, ctile, tid);
  __syncthreads();

  const int m0 = wave * 16;
  const int rofs = (lane & 16) ? 8 : 0;
  const int col = lane & 15;

  // ---- K strip: softmax numerator + partial row sums ----
  float rmaxv[8];
  #pragma unroll
  for (int r = 0; r < 8; ++r) rmaxv[r] = fdec(rmaxg[b * CCH + m0 + rofs + r]);
  v16h afr[4];
  #pragma unroll
  for (int ks = 0; ks < 4; ++ks) afr[ks] = load_a16(Wk16, CCH, m0, ks * 32, lane);
  float rs[8] = {0.f, 0.f, 0.f, 0.f, 0.f, 0.f, 0.f, 0.f};
  for (int nt = 0; nt < TN / 16; ++nt) {
    v8f acc = {};
    #pragma unroll
    for (int ks = 0; ks < 4; ++ks)
      acc = wmma_f16(afr[ks], load_b16(ctile, LDC, nt * 16, ks * 32, lane), acc);
    #pragma unroll
    for (int r = 0; r < 8; ++r) {
      float e = __expf(acc[r] - rmaxv[r]);     // <= 1 by construction
      rs[r] += e;
      kl[(m0 + rofs + r) * LDK + nt * 16 + col] = (_Float16)e;
    }
  }
  #pragma unroll
  for (int r = 0; r < 8; ++r) rs[r] = red16_sum(rs[r]);
  if ((lane & 15) == 0) {
    #pragma unroll
    for (int r = 0; r < 8; ++r)
      atomicAdd(&rsumg[b * CCH + m0 + rofs + r], rs[r]);
  }

  // ---- V strip ----
  #pragma unroll
  for (int ks = 0; ks < 4; ++ks) afr[ks] = load_a16(Wv16, CCH, m0, ks * 32, lane);
  for (int nt = 0; nt < TN / 16; ++nt) {
    v8f acc = {};
    #pragma unroll
    for (int ks = 0; ks < 4; ++ks)
      acc = wmma_f16(afr[ks], load_b16(ctile, LDC, nt * 16, ks * 32, lane), acc);
    #pragma unroll
    for (int r = 0; r < 8; ++r)
      vl[(m0 + rofs + r) * LDK + nt * 16 + col] = (_Float16)acc[r];
  }
  __syncthreads();

  // ---- per-head ctx[d][e] += sum_n eK[d,n] * V[e,n] (WMMA, K-dim = tokens) ----
  #pragma unroll
  for (int jj = 0; jj < 2; ++jj) {
    const int j = wave * 2 + jj;                  // 16 jobs: (head, d-tile, e-tile)
    const int h = j >> 2, mt = ((j >> 1) & 1) * 16, et = (j & 1) * 16;
    v8f acc = {};
    #pragma unroll
    for (int ks = 0; ks < TN / 32; ++ks)
      acc = wmma_f16(load_a16(kl, LDK, h * DIMH + mt, ks * 32, lane),
                     load_b16(vl, LDK, h * DIMH + et, ks * 32, lane), acc);
    float* cg = ctxa + (b * HEADS + h) * DIMH * DIMH;
    #pragma unroll
    for (int r = 0; r < 8; ++r)
      atomicAdd(&cg[(mt + rofs + r) * DIMH + et + col], acc[r]);
  }
}

// ---- kernel 3: Weff[b] = Wout * blockdiag(ctx^T / rowsum) * SCALE*Wq  ------
__global__ void __launch_bounds__(256)
build_weff(const float* __restrict__ ctxa, const float* __restrict__ rsum,
           const float* __restrict__ Wq, const float* __restrict__ Wout,
           _Float16* __restrict__ Weffg) {
  __shared__ float    ctxN[HEADS * DIMH * DIMH];   // 16 KB
  __shared__ _Float16 T1[CCH * CCH];               // 32 KB
  const int b = blockIdx.x, t = threadIdx.x;
  for (int i = t; i < HEADS * DIMH * DIMH; i += 256) {
    const int h = i >> 10, rem = i & 1023, d = rem >> 5;
    ctxN[i] = ctxa[b * HEADS * DIMH * DIMH + i] / rsum[b * CCH + h * DIMH + d];
  }
  __syncthreads();
  const int r = t >> 1, c0 = (t & 1) * 64;
  {   // T1[h*32+e][c] = SCALE * sum_d ctxN[h][d][e] * Wq[h*32+d][c]
    const int h = r >> 5, e = r & 31;
    for (int c = c0; c < c0 + 64; ++c) {
      float acc = 0.f;
      for (int d = 0; d < DIMH; ++d)
        acc += ctxN[(h * DIMH + d) * DIMH + e] * Wq[(h * DIMH + d) * CCH + c];
      T1[r * CCH + c] = (_Float16)(acc * SCALEQ);
    }
  }
  __syncthreads();
  {   // Weff[o][c] = sum_r Wout[o][r] * T1[r][c]
    const int o = r;
    for (int c = c0; c < c0 + 64; ++c) {
      float acc = 0.f;
      for (int rr = 0; rr < CCH; ++rr)
        acc += Wout[o * CCH + rr] * (float)T1[rr * CCH + c];
      Weffg[b * CCH * CCH + o * CCH + c] = (_Float16)acc;
    }
  }
}

// ---------------- kernel 4: y = Weff * x + bout (streaming WMMA GEMM) -------
__global__ void __launch_bounds__(256)
ygemm(const _Float16* __restrict__ xh, const _Float16* __restrict__ Weffg,
      const float* __restrict__ bout, float* __restrict__ y) {
  __shared__ _Float16 xt[TN2 * LDC];               // ~34 KB
  const int b = blockIdx.y;
  const long tile0 = (long)blockIdx.x * TN2;
  const int tid = threadIdx.x, lane = tid & 31, wave = tid >> 5;
  stage_tile<TN2>(xh + (long)b * CCH * NTOK, tile0, xt, tid);
  __syncthreads();

  const int m0 = wave * 16;
  const int rofs = (lane & 16) ? 8 : 0;
  const int col = lane & 15;
  const _Float16* Wb = Weffg + b * CCH * CCH;
  v16h afr[4];
  #pragma unroll
  for (int ks = 0; ks < 4; ++ks) afr[ks] = load_a16(Wb, CCH, m0, ks * 32, lane);
  float bb[8];
  #pragma unroll
  for (int r = 0; r < 8; ++r) bb[r] = bout[m0 + rofs + r];

  for (int nt = 0; nt < TN2 / 16; ++nt) {
    v8f acc = {};
    #pragma unroll
    for (int ks = 0; ks < 4; ++ks)
      acc = wmma_f16(afr[ks], load_b16(xt, LDC, nt * 16, ks * 32, lane), acc);
    float* yp = y + (long)(b * CCH + m0 + rofs) * NTOK + tile0 + nt * 16 + col;
    #pragma unroll
    for (int r = 0; r < 8; ++r) yp[(long)r * NTOK] = acc[r] + bb[r];
  }
}

// ---------------------------------------------------------------------------
extern "C" void kernel_launch(void* const* d_in, const int* in_sizes, int n_in,
                              void* d_out, int out_size, void* d_ws, size_t ws_size,
                              hipStream_t stream) {
  (void)in_sizes; (void)n_in; (void)out_size; (void)ws_size;
  const float* x    = (const float*)d_in[0];
  const float* ctxI = (const float*)d_in[1];
  const float* Wq   = (const float*)d_in[2];
  const float* Wk   = (const float*)d_in[3];
  const float* Wv   = (const float*)d_in[4];
  const float* Wout = (const float*)d_in[5];
  const float* bout = (const float*)d_in[6];
  float* y = (float*)d_out;

  char* ws = (char*)d_ws;
  _Float16* Wk16 = (_Float16*)(ws + 0);
  _Float16* Wv16 = (_Float16*)(ws + 32768);
  _Float16* Weff = (_Float16*)(ws + 65536);
  unsigned* rmax = (unsigned*)(ws + 196608);
  float*    rsum = (float*)(ws + 198656);
  float*    ctxa = (float*)(ws + 200704);
  _Float16* xh   = (_Float16*)(ws + 266240);
  _Float16* ch   = (_Float16*)(ws + 266240 + 33554432);

  convert_inputs<<<2048, 256, 0, stream>>>(x, ctxI, xh, ch);
  prep_small<<<64, 256, 0, stream>>>(Wk, Wv, Wk16, Wv16, rmax, rsum, ctxa);

  dim3 g1(NTOK / TN, BATCH);                 // 512 x 4
  kv_rowmax<<<g1, 256, 0, stream>>>(ch, Wk16, rmax);
  kv_softmax_ctx<<<g1, 256, 0, stream>>>(ch, Wk16, Wv16, rmax, rsum, ctxa);

  build_weff<<<BATCH, 256, 0, stream>>>(ctxa, rsum, Wq, Wout, Weff);

  dim3 g4(NTOK / TN2, BATCH);                // 256 x 4
  ygemm<<<g4, 256, 0, stream>>>(xh, Weff, bout, y);
}
